// single_conv_up_52218212385037
// MI455X (gfx1250) — matrix operand order repaired and checked by reading, not compiled, not run on hardware
//
#include <hip/hip_runtime.h>
#include <hip/hip_bf16.h>

// ---------------------------------------------------------------------------
// CondConv3d(3x3x3, SAME) + BatchNorm3d(batch stats) + LeakyReLU(0.1)
// + nearest upsample x2, for MI455X (gfx1250, wave32, WMMA).
//
// Implicit-GEMM conv: M = 2x16 spatial positions per wave, N = 16 out-channels
// (x2 tiles for Cout=32), K = 32 in-channels per kernel tap, 27 taps.
// v_wmma_f32_16x16x32_bf16, fp32 accumulate; weights staged into LDS with
// global_load_async_to_lds_b64 (ASYNCcnt) and read as B fragments via DS.
// ---------------------------------------------------------------------------

typedef __attribute__((ext_vector_type(16))) __bf16 v16bf;
typedef __attribute__((ext_vector_type(8)))  __bf16 v8bf;
typedef __attribute__((ext_vector_type(8)))  float  v8f;

#define BATCH   2
#define CIN     32
#define COUT    32
#define DIM     40
#define NPOS    64000          // 40^3
#define NEXP    3
#define NCHUNK  27             // 3x3x3 taps
#define PK_PER_B (NCHUNK * 2 * 32 * 16)   // 27648 bf16 elems / sample
#define NOUTPOS  512000        // 80^3

// ---------------------------------------------------------------------------
// Kernel A: routing + combined bias + WMMA-layout bf16 weight packing.
// B-fragment layout (16x16x32 bf16): lane l (0-15) holds K={0..7,16..23} at
// N=l; lane l (16-31) holds K={8..15,24..31} at N=l-16.  Packed as
// pk[b][chunk][ntile][lane][j] so each lane reads 32 contiguous bytes.
// ---------------------------------------------------------------------------
__global__ void __launch_bounds__(256)
condconv_prep(const float* __restrict__ emb, const float* __restrict__ rw,
              const float* __restrict__ rb,  const float* __restrict__ ek,
              const float* __restrict__ ebias,
              float* __restrict__ gsum, float* __restrict__ gsq,
              float* __restrict__ bb, __bf16* __restrict__ pk)
{
    const int tid = threadIdx.x;
    if (tid < COUT) { gsum[tid] = 0.0f; gsq[tid] = 0.0f; }

    // routing: r[b][e] = sigmoid(emb[b] * rw[e] + rb[e])   (EMB == 1)
    float r[BATCH][NEXP];
    #pragma unroll
    for (int b = 0; b < BATCH; ++b)
        #pragma unroll
        for (int e = 0; e < NEXP; ++e) {
            float t = emb[b] * rw[e] + rb[e];
            r[b][e] = 1.0f / (1.0f + __expf(-t));
        }

    if (tid < BATCH * COUT) {
        int b = tid >> 5, o = tid & 31;
        float s = 0.0f;
        #pragma unroll
        for (int e = 0; e < NEXP; ++e) s += r[b][e] * ebias[e * COUT + o];
        bb[tid] = s;
    }

    for (int idx = tid; idx < BATCH * PK_PER_B; idx += blockDim.x) {
        int j    = idx & 15;
        int rest = idx >> 4;
        int l    = rest & 31; rest >>= 5;
        int t    = rest & 1;  rest >>= 1;
        int c    = rest % NCHUNK;
        int b    = rest / NCHUNK;
        int i    = ((l >> 4) << 3) + (j & 7) + ((j >> 3) << 4);     // K row = input channel
        int o    = (t << 4) + (l & 15);     // N col = output channel
        float s = 0.0f;
        #pragma unroll
        for (int e = 0; e < NEXP; ++e)
            s += r[b][e] * ek[(((e * COUT + o) * CIN + i) * NCHUNK) + c];
        pk[idx] = (__bf16)s;
    }
}

// ---------------------------------------------------------------------------
// Kernel B: x [b][i][p] fp32  ->  channel-last bf16 x_cl [b][p][i]
// via LDS transpose (coalesced reads AND writes).
// ---------------------------------------------------------------------------
__global__ void __launch_bounds__(256)
x_to_bf16(const float* __restrict__ x, __bf16* __restrict__ xcl)
{
    __shared__ __bf16 tile[64][CIN + 1];       // +1 pad: avoid bank conflicts
    const int b  = blockIdx.x / 1000;
    const int p0 = (blockIdx.x % 1000) * 64;
    for (int idx = threadIdx.x; idx < CIN * 64; idx += blockDim.x) {
        int i = idx / 64, t = idx % 64;
        tile[t][i] = (__bf16)x[((size_t)(b * CIN + i)) * NPOS + p0 + t];
    }
    __syncthreads();
    for (int idx = threadIdx.x; idx < 64 * CIN; idx += blockDim.x) {
        int t = idx >> 5, i = idx & 31;
        xcl[((size_t)(b * NPOS + p0 + t)) * CIN + i] = tile[t][i];
    }
}

// ---------------------------------------------------------------------------
// Kernel C: WMMA implicit-GEMM conv + bias; y fp32 [b][o][p]; per-channel
// sum / sum^2 via LDS ds_add_f32 then one global_atomic_add_f32 per block.
// 8 waves/block, 2 M-tiles (32 positions) per wave -> 500 blocks.
// Weights: async global->LDS staging (ASYNCcnt), B fragments via ds_load.
// ---------------------------------------------------------------------------
__global__ void __launch_bounds__(256)
condconv_wmma(const __bf16* __restrict__ xcl, const __bf16* __restrict__ pk,
              const float* __restrict__ bb, float* __restrict__ y,
              float* __restrict__ gsum, float* __restrict__ gsq)
{
    __shared__ __attribute__((aligned(32))) __bf16 wlds[PK_PER_B]; // 55296 B
    __shared__ float lsum[COUT], lsq[COUT];

    const int tid = threadIdx.x;
    const int b   = blockIdx.x / 250;          // all 8 waves share one sample

    if (tid < COUT) { lsum[tid] = 0.0f; lsq[tid] = 0.0f; }

    // --- async stage this sample's packed weights into LDS ----------------
    // global_load_async_to_lds_b64: vdst = LDS byte address, vaddr = 64b ptr.
    {
        const char* src = reinterpret_cast<const char*>(pk + (size_t)b * PK_PER_B);
        unsigned ldsbase = (unsigned)(size_t)(&wlds[0]);
        #pragma unroll 1
        for (int i = tid; i < PK_PER_B / 4; i += 256) {   // 6912 x b64
            unsigned          loff = ldsbase + i * 8u;
            unsigned long long ga  = (unsigned long long)(size_t)(src + i * 8);
            asm volatile("global_load_async_to_lds_b64 %0, %1, off"
                         :: "v"(loff), "v"(ga) : "memory");
        }
        asm volatile("s_wait_asynccnt 0" ::: "memory");
    }
    __syncthreads();

    const int wave  = tid >> 5;
    const int lane  = tid & 31;
    const int tp    = blockIdx.x * 8 + wave;     // wave-pair index, 0..3999
    const int pbase = (tp % 2000) << 5;          // 32 positions per wave
    const int m     = lane & 15;

    const int pos0 = pbase + m;
    const int pos1 = pos0 + 16;
    const int d0 = pos0 / 1600, r0 = pos0 - d0 * 1600;
    const int h0 = r0 / 40,     w0 = r0 - h0 * 40;
    const int d1 = pos1 / 1600, r1 = pos1 - d1 * 1600;
    const int h1 = r1 / 40,     w1 = r1 - h1 * 40;

    const int half = lane >> 4;                  // 0 | 1
    const int c0   = half << 3;                  // A-frag channel base: 0 | 8

    const int   o0 = lane & 15;
    const int   o1 = 16 + o0;
    const float bias0 = bb[b * COUT + o0];
    const float bias1 = bb[b * COUT + o1];
    v8f acc00, acc01, acc10, acc11;
    #pragma unroll
    for (int j = 0; j < 8; ++j) {
        acc00[j] = bias0; acc01[j] = bias1;
        acc10[j] = bias0; acc11[j] = bias1;
    }

    const __bf16* __restrict__ xb = xcl + (size_t)b * NPOS * CIN;
    __builtin_prefetch(xb + (size_t)pos0 * CIN, 0, 0);  // global_prefetch_b8
    __builtin_prefetch(xb + (size_t)pos1 * CIN, 0, 0);

    #pragma unroll 1
    for (int kdh = 0; kdh < 9; ++kdh) {
        const int kd = kdh / 3 - 1;              // -1..1
        const int kh = kdh % 3 - 1;
        const bool vdh0 = ((unsigned)(d0 + kd) < 40u) & ((unsigned)(h0 + kh) < 40u);
        const bool vdh1 = ((unsigned)(d1 + kd) < 40u) & ((unsigned)(h1 + kh) < 40u);
        const int np0 = pos0 + kd * 1600 + kh * 40;
        const int np1 = pos1 + kd * 1600 + kh * 40;

        #pragma unroll
        for (int kw = 0; kw < 3; ++kw) {
            const int c = kdh * 3 + kw;

            // A fragments: lanes 0-15 hold K={0..7,16..23},
            //              lanes 16-31 hold K={8..15,24..31}
            v16bf a0 = {}, a1 = {};
            if (vdh0 & ((unsigned)(w0 + kw - 1) < 40u)) {
                const __bf16* xp = xb + (size_t)(np0 + kw - 1) * CIN + c0;
                const v8bf lo = *reinterpret_cast<const v8bf*>(xp);
                const v8bf hc = *reinterpret_cast<const v8bf*>(xp + 16);
                #pragma unroll
                for (int j = 0; j < 8; ++j) { a0[j] = lo[j]; a0[8 + j] = hc[j]; }
            }
            if (vdh1 & ((unsigned)(w1 + kw - 1) < 40u)) {
                const __bf16* xp = xb + (size_t)(np1 + kw - 1) * CIN + c0;
                const v8bf lo = *reinterpret_cast<const v8bf*>(xp);
                const v8bf hc = *reinterpret_cast<const v8bf*>(xp + 16);
                #pragma unroll
                for (int j = 0; j < 8; ++j) { a1[j] = lo[j]; a1[8 + j] = hc[j]; }
            }

            // B fragments from LDS: 32 contiguous bytes per lane
            const v16bf bw0 = *reinterpret_cast<const v16bf*>(
                &wlds[((c * 2 + 0) * 32 + lane) * 16]);
            const v16bf bw1 = *reinterpret_cast<const v16bf*>(
                &wlds[((c * 2 + 1) * 32 + lane) * 16]);

            acc00 = __builtin_amdgcn_wmma_f32_16x16x32_bf16(
                false, a0, false, bw0, (short)0, acc00, false, false);
            acc01 = __builtin_amdgcn_wmma_f32_16x16x32_bf16(
                false, a0, false, bw1, (short)0, acc01, false, false);
            acc10 = __builtin_amdgcn_wmma_f32_16x16x32_bf16(
                false, a1, false, bw0, (short)0, acc10, false, false);
            acc11 = __builtin_amdgcn_wmma_f32_16x16x32_bf16(
                false, a1, false, bw1, (short)0, acc11, false, false);
        }
    }

    // C/D layout: VGPR j holds M=j (lanes 0-15) / M=j+8 (lanes 16-31), N=lane&15
    float s0 = 0.f, q0 = 0.f, s1 = 0.f, q1 = 0.f;
    float* __restrict__ y0 = y + ((size_t)(b * COUT + o0)) * NPOS;
    float* __restrict__ y1 = y + ((size_t)(b * COUT + o1)) * NPOS;
    const int mofs = half << 3;
    #pragma unroll
    for (int j = 0; j < 8; ++j) {
        const int pA = pbase + mofs + j;         // M-tile 0
        const int pB = pA + 16;                  // M-tile 1
        const float vA0 = acc00[j], vA1 = acc01[j];
        const float vB0 = acc10[j], vB1 = acc11[j];
        y0[pA] = vA0; y1[pA] = vA1;
        y0[pB] = vB0; y1[pB] = vB1;
        s0 += vA0 + vB0; q0 += vA0 * vA0 + vB0 * vB0;
        s1 += vA1 + vB1; q1 += vA1 * vA1 + vB1 * vB1;
    }
    atomicAdd(&lsum[o0], s0); atomicAdd(&lsq[o0], q0);   // ds_add_f32
    atomicAdd(&lsum[o1], s1); atomicAdd(&lsq[o1], q1);
    __syncthreads();
    if (tid < COUT) {
        unsafeAtomicAdd(&gsum[tid], lsum[tid]);          // global_atomic_add_f32
        unsafeAtomicAdd(&gsq[tid],  lsq[tid]);
    }
}

// ---------------------------------------------------------------------------
// Kernel D: fold batch stats + gamma/beta into per-channel scale/shift.
// ---------------------------------------------------------------------------
__global__ void bn_finalize(const float* __restrict__ gsum,
                            const float* __restrict__ gsq,
                            const float* __restrict__ gamma,
                            const float* __restrict__ beta,
                            float* __restrict__ scsh)
{
    const int o = threadIdx.x;
    if (o < COUT) {
        const float invN = 1.0f / (float)(BATCH * NPOS);
        const float mean = gsum[o] * invN;
        const float var  = gsq[o] * invN - mean * mean;
        const float sc   = gamma[o] * rsqrtf(var + 1e-5f);
        scsh[o]        = sc;
        scsh[COUT + o] = beta[o] - mean * sc;
    }
}

// ---------------------------------------------------------------------------
// Kernel E: normalize + LeakyReLU(0.1) + nearest upsample x2.
// One thread per (b,o,p); 4x float2 stores cover the 2x2x2 block.
// This 131 MB store stream is the roofline-dominant stage.
// ---------------------------------------------------------------------------
__global__ void __launch_bounds__(256)
norm_act_upsample(const float* __restrict__ y, const float* __restrict__ scsh,
                  float* __restrict__ out)
{
    const size_t g = (size_t)blockIdx.x * blockDim.x + threadIdx.x;  // < 4.096M
    const int p    = (int)(g % NPOS);
    const int rest = (int)(g / NPOS);     // b*32 + o
    const int o    = rest & 31;

    float v = y[g] * scsh[o] + scsh[COUT + o];
    v = (v >= 0.0f) ? v : 0.1f * v;

    const int d  = p / 1600;
    const int hw = p - d * 1600;
    const int h  = hw / 40;
    const int w  = hw - h * 40;

    float2 pr; pr.x = v; pr.y = v;
    float2* __restrict__ o2 = reinterpret_cast<float2*>(out);
    const size_t base2 = (size_t)rest * (NOUTPOS / 2) +
                         (size_t)d * 6400 + (size_t)h * 80 + w;
    o2[base2]        = pr;   // (2d,   2h,   2w..2w+1)
    o2[base2 + 40]   = pr;   // (2d,   2h+1, ...)
    o2[base2 + 3200] = pr;   // (2d+1, 2h,   ...)
    o2[base2 + 3240] = pr;   // (2d+1, 2h+1, ...)
}

// ---------------------------------------------------------------------------
extern "C" void kernel_launch(void* const* d_in, const int* in_sizes, int n_in,
                              void* d_out, int out_size, void* d_ws, size_t ws_size,
                              hipStream_t stream)
{
    const float* x     = (const float*)d_in[0];
    const float* emb   = (const float*)d_in[1];
    const float* rw    = (const float*)d_in[2];
    const float* rb    = (const float*)d_in[3];
    const float* ek    = (const float*)d_in[4];
    const float* eb    = (const float*)d_in[5];
    const float* gamma = (const float*)d_in[6];
    const float* beta  = (const float*)d_in[7];

    char* ws = (char*)d_ws;
    float*  gsum = (float*)(ws + 0);          //  32 f
    float*  gsq  = (float*)(ws + 128);        //  32 f
    float*  bb   = (float*)(ws + 256);        //  64 f
    float*  scsh = (float*)(ws + 512);        //  64 f
    __bf16* pk   = (__bf16*)(ws + 1024);      //  55296 bf16  -> ends 111616
    __bf16* xcl  = (__bf16*)(ws + 111616);    //  4.096M bf16 -> ends 8303616
    float*  y    = (float*)(ws + 8303616);    //  4.096M f32  -> ends ~24.7 MB
    float*  out  = (float*)d_out;

    condconv_prep    <<<1,     256, 0, stream>>>(emb, rw, rb, ek, eb,
                                                 gsum, gsq, bb, pk);
    x_to_bf16        <<<2000,  256, 0, stream>>>(x, xcl);
    condconv_wmma    <<<500,   256, 0, stream>>>(xcl, pk, bb, y, gsum, gsq);
    bn_finalize      <<<1,      32, 0, stream>>>(gsum, gsq, gamma, beta, scsh);
    norm_act_upsample<<<16000, 256, 0, stream>>>(y, scsh, out);
}
